// BatchConv2DLayer_3178275799256
// MI455X (gfx1250) — matrix operand order repaired and checked by reading, not compiled, not run on hardware
//
#include <hip/hip_runtime.h>

typedef __attribute__((ext_vector_type(16))) _Float16 v16h;
typedef __attribute__((ext_vector_type(8)))  _Float16 v8h;
typedef __attribute__((ext_vector_type(8)))  float    v8f;
typedef __attribute__((ext_vector_type(4)))  float    v4f;

#define BB   32
#define CC   32
#define HH   256
#define WW   256

// LDS geometry (half-element strides, chosen so every vector access is 16B aligned)
#define XCOLS 258   // padded width: 256 + left/right zero columns
#define XCS   40    // channel stride (32 padded to 40 -> 80B per column, 16B aligned)
#define WROW  296   // per-c_out weight row stride (288 padded, 592B, 16B aligned)

__global__ __launch_bounds__(256)
void batch_conv3x3_wmma(const float* __restrict__ x,
                        const float* __restrict__ wgt,
                        const float* __restrict__ bias,
                        float* __restrict__ out) {
  __shared__ _Float16 xs[3 * XCOLS * XCS];  // 61,920 B: rows h-1..h+1, [row][pcol][c]
  __shared__ _Float16 ws[32 * WROW];        // 18,944 B: [c_out][tap][c_in]

  const int h   = blockIdx.x;   // output row
  const int b   = blockIdx.y;   // batch
  const int tid = threadIdx.x;  // 0..255 (8 wave32)

  // ---- Stage 3 input rows (f32 -> f16), transposed to channel-contiguous ----
  // 3 rows x 32 ch x 256 cols = 6144 float4 chunks; 24 per thread, b128 coalesced.
  #pragma unroll 4
  for (int i = 0; i < 24; ++i) {
    int q   = i * 256 + tid;      // 0..6143
    int dr  = q >> 11;            // /2048: 0..2
    int rem = q & 2047;
    int c   = rem >> 6;           // /64:   0..31
    int w4  = rem & 63;           // float4 index within the row
    int hin = h + dr - 1;
    v4f v = {0.f, 0.f, 0.f, 0.f};
    if (hin >= 0 && hin < HH)
      v = *(const v4f*)&x[(((size_t)b * CC + c) * HH + hin) * WW + w4 * 4];
    int base = (dr * XCOLS + w4 * 4 + 1) * XCS + c;
    xs[base]           = (_Float16)v.x;
    xs[base + XCS]     = (_Float16)v.y;
    xs[base + 2 * XCS] = (_Float16)v.z;
    xs[base + 3 * XCS] = (_Float16)v.w;
  }
  if (tid < 96) {                           // zero the left/right halo columns
    int dr = tid >> 5;
    int c  = tid & 31;
    xs[(dr * XCOLS + 0)   * XCS + c] = (_Float16)0.f;
    xs[(dr * XCOLS + 257) * XCS + c] = (_Float16)0.f;
  }
  // ---- Stage this sample's weights as [m][tap][c_in] f16 ----
  // 32*288 = 9216 floats = 2304 float4; 9 per thread, b128 coalesced.
  #pragma unroll
  for (int i = 0; i < 9; ++i) {
    int e0 = (i * 256 + tid) * 4;                    // first flat element of chunk
    v4f v = *(const v4f*)&wgt[(size_t)b * 32 * 288 + e0];
    #pragma unroll
    for (int k = 0; k < 4; ++k) {
      int e   = e0 + k;                              // flat (m, c, kk) index
      int m   = e / 288;
      int rem = e % 288;
      int c   = rem / 9;
      int kk  = rem % 9;                             // kk = r*3+s (OIHW flat order)
      ws[m * WROW + kk * 32 + c] = (_Float16)v[k];
    }
  }
  __syncthreads();

  const int wave = tid >> 5;
  const int lane = tid & 31;
  const int n    = lane & 15;
  const int hi   = lane >> 4;               // which half of the wave
  const int t0   = wave * 2;                // this wave's first W-tile (of 16)

  // Per-lane LDS bases matching the CDNA5 WMMA VGPR layouts.
  // A 16x32 f16: lane<16 -> K {0..7,16..23}; lane>=16 -> K {8..15,24..31}
  const int a0base = n * WROW + hi * 8;          // c_out tile 0 (M = n)
  const int a1base = (16 + n) * WROW + hi * 8;   // c_out tile 1 (M = 16+n)
  // B 32x16 f16: lane<16 -> K {0..7,16..23} at col n; lane>=16 -> K {8..15,24..31}
  const int b0col = t0 * 16 + n;                 // padded col = w + s
  const int bch   = hi * 8;

  v8f acc[4];
  {
    // C/D layout: VGPR j holds M = j + 8*hi; seed with bias
    const float* bp = bias + (size_t)b * 32;
    #pragma unroll
    for (int j = 0; j < 8; ++j) {
      float v0 = bp[hi * 8 + j];
      float v1 = bp[16 + hi * 8 + j];
      acc[0][j] = v0; acc[1][j] = v1;   // W-tile t0
      acc[2][j] = v0; acc[3][j] = v1;   // W-tile t0+1
    }
  }

  #pragma unroll
  for (int kk = 0; kk < 9; ++kk) {      // 9 taps x K=32 channels = full 288 reduction
    const int r = kk / 3, s = kk % 3;

    v8h a0lo = *(const v8h*)&ws[a0base + kk * 32];
    v8h a0hi = *(const v8h*)&ws[a0base + kk * 32 + 16];
    v8h a1lo = *(const v8h*)&ws[a1base + kk * 32];
    v8h a1hi = *(const v8h*)&ws[a1base + kk * 32 + 16];
    v16h a0 = __builtin_shufflevector(a0lo, a0hi, 0,1,2,3,4,5,6,7,8,9,10,11,12,13,14,15);
    v16h a1 = __builtin_shufflevector(a1lo, a1hi, 0,1,2,3,4,5,6,7,8,9,10,11,12,13,14,15);

    int p0 = (r * XCOLS + b0col + s) * XCS + bch;          // two 16B chunks, 32B apart
    int p1 = (r * XCOLS + b0col + 16 + s) * XCS + bch;
    v8h b0lo = *(const v8h*)&xs[p0];
    v8h b0hi = *(const v8h*)&xs[p0 + 16];
    v8h b1lo = *(const v8h*)&xs[p1];
    v8h b1hi = *(const v8h*)&xs[p1 + 16];
    v16h bt0 = __builtin_shufflevector(b0lo, b0hi, 0,1,2,3,4,5,6,7,8,9,10,11,12,13,14,15);
    v16h bt1 = __builtin_shufflevector(b1lo, b1hi, 0,1,2,3,4,5,6,7,8,9,10,11,12,13,14,15);

    acc[0] = __builtin_amdgcn_wmma_f32_16x16x32_f16(false, a0, false, bt0, (short)0, acc[0], false, false);
    acc[1] = __builtin_amdgcn_wmma_f32_16x16x32_f16(false, a1, false, bt0, (short)0, acc[1], false, false);
    acc[2] = __builtin_amdgcn_wmma_f32_16x16x32_f16(false, a0, false, bt1, (short)0, acc[2], false, false);
    acc[3] = __builtin_amdgcn_wmma_f32_16x16x32_f16(false, a1, false, bt1, (short)0, acc[3], false, false);
  }

  // ---- Store D non-temporally: output is streamed once, keep x rows in L2 ----
  const int w0 = t0 * 16 + n;
  #pragma unroll
  for (int j = 0; j < 8; ++j) {
    int m = hi * 8 + j;
    size_t o0 = (((size_t)b * 32 + m)      * HH + h) * WW + w0;
    size_t o1 = (((size_t)b * 32 + 16 + m) * HH + h) * WW + w0;
    __builtin_nontemporal_store(acc[0][j], &out[o0]);
    __builtin_nontemporal_store(acc[1][j], &out[o1]);
    __builtin_nontemporal_store(acc[2][j], &out[o0 + 16]);
    __builtin_nontemporal_store(acc[3][j], &out[o1 + 16]);
  }
}

extern "C" void kernel_launch(void* const* d_in, const int* in_sizes, int n_in,
                              void* d_out, int out_size, void* d_ws, size_t ws_size,
                              hipStream_t stream) {
  const float* x    = (const float*)d_in[0];  // [32,32,256,256] f32
  const float* wgt  = (const float*)d_in[1];  // [32,32,32,3,3]  f32
  const float* bias = (const float*)d_in[2];  // [32,32]         f32
  float* out        = (float*)d_out;          // [32,32,256,256] f32

  dim3 grid(HH, BB);   // one workgroup per (output row, batch)
  batch_conv3x3_wmma<<<grid, 256, 0, stream>>>(x, wgt, bias, out);
}
